// Transformer_87771951661552
// MI455X (gfx1250) — matrix operand order repaired and checked by reading, not compiled, not run on hardware
//
#include <hip/hip_runtime.h>
#include <hip/hip_bf16.h>

typedef unsigned short     u16;
typedef unsigned int       u32;
typedef unsigned long long u64;

typedef __attribute__((ext_vector_type(16))) __bf16 v16bf;
typedef __attribute__((ext_vector_type(8)))  float  v8f;
typedef __attribute__((ext_vector_type(4)))  u32    u32x4;
typedef __attribute__((ext_vector_type(8)))  u32    u32x8;

// ---------------------------------------------------------------- constants
#define BB   4
#define TT   2048
#define EE   1024
#define NHH  16
#define DHH  64
#define NEE  2048
#define HIDD 16
#define MTOK 8192            // B*T
#define E3   3072

// ---------------------------------------------------------------- bf16 utils
__device__ __forceinline__ u16 f2bf(float f) {
    union { float f; u32 u; } v; v.f = f;
    u32 r = v.u + 0x7FFFu + ((v.u >> 16) & 1u);   // round-to-nearest-even
    return (u16)(r >> 16);
}
__device__ __forceinline__ float bf2f(u16 h) {
    union { u32 u; float f; } v; v.u = ((u32)h) << 16;
    return v.f;
}

union Bf16x16 {
    v16bf v;
    u16   s[16];
    u32   u[8];
    uint4 q[2];
};

__device__ __forceinline__ v8f vzero8() {
    v8f v;
#pragma unroll
    for (int i = 0; i < 8; ++i) v[i] = 0.0f;
    return v;
}

__device__ __forceinline__ v8f wmma_bf16(v16bf a, v16bf b, v8f c) {
    return __builtin_amdgcn_wmma_f32_16x16x32_bf16(false, a, false, b, (short)0, c, false, false);
}

// XOR-butterfly lane exchange via DS_SWIZZLE_B32 (group-of-32 mode:
// offset[15]=0, xor_mask=offset[14:10], or=0, and_mask=0x1F). One DS op,
// no index math / bounds compare like __shfl_xor. EXEC is all-ones here.
template <int XORMASK>
__device__ __forceinline__ float swz_xor(float v) {
    union { float f; int i; } u; u.f = v;
    u.i = __builtin_amdgcn_ds_swizzle(u.i, (XORMASK << 10) | 0x1F);
    return u.f;
}
__device__ __forceinline__ float red_max16(float v) {
    v = fmaxf(v, swz_xor<8>(v));
    v = fmaxf(v, swz_xor<4>(v));
    v = fmaxf(v, swz_xor<2>(v));
    v = fmaxf(v, swz_xor<1>(v));
    return v;
}
__device__ __forceinline__ float red_sum16(float v) {
    v += swz_xor<8>(v);
    v += swz_xor<4>(v);
    v += swz_xor<2>(v);
    v += swz_xor<1>(v);
    return v;
}

// A fragment (16x32 bf16, MxK), source row-major with rowStride (elements).
// ISA layout: lane m = lane&15, half = lane>>4; VGPR v holds K pair
// k = (v<4?0:16) + half*8 + 2*(v&3)
__device__ __forceinline__ v16bf load_a_frag(const u16* base, int rowStride) {
    int lane = threadIdx.x & 31;
    int m = lane & 15, half = lane >> 4;
    Bf16x16 r;
    const u16* row = base + (size_t)m * rowStride;
#pragma unroll
    for (int v = 0; v < 8; ++v) {
        int k = ((v < 4) ? 0 : 16) + half * 8 + (v & 3) * 2;
        r.u[v] = *(const u32*)(row + k);
    }
    return r.v;
}

// B fragment (32x16 bf16, KxN), source stored N-major: tile[n*ldk + k].
// ISA layout: lane n = lane&15, half = lane>>4; elements i -> k = (i<8?0:16) + half*8 + (i&7)
__device__ __forceinline__ v16bf load_b_frag(const u16* tile, int ldk) {
    int lane = threadIdx.x & 31;
    int n = lane & 15, half = lane >> 4;
    Bf16x16 r;
    const u16* p = tile + (size_t)n * ldk + half * 8;
    r.q[0] = *(const uint4*)(p);
    r.q[1] = *(const uint4*)(p + 16);
    return r.v;
}

// ---------------------------------------------------------------- TDM
// Tensor Data Mover: DMA a 2D bf16 tile (rows x elems_per_row, row stride in
// elements) from global memory into LDS, per CDNA5 ISA
// D# group0/group1 layout. Issue once per wave; track with TENSORcnt.
__device__ __forceinline__ void tdm_load_2d_bf16(u32 lds_byte_addr,
                                                 const void* gaddr,
                                                 u32 elems_per_row,   // tile_dim0
                                                 u32 rows,            // tile_dim1
                                                 u32 row_stride_el,   // tensor_dim0_stride
                                                 u32 tensor_d0,
                                                 u32 tensor_d1) {
    union { u32x4 v; u32 w[4]; } g0;
    union { u32x8 v; u32 w[8]; } g1;
    u64 ga = (u64)(uintptr_t)gaddr;
    // group0: [1:0]=count=1, [63:32]=lds_addr, [120:64]=global_addr, [127:126]=type=2
    g0.w[0] = 1u;
    g0.w[1] = lds_byte_addr;
    g0.w[2] = (u32)(ga & 0xFFFFFFFFu);
    g0.w[3] = (u32)((ga >> 32) & 0x01FFFFFFu) | (2u << 30);
    // group1: [15:0]=wg_mask=0, [17:16]=data_size=1 (2B), rest of flags 0
    g1.w[0] = (1u << 16);
    // [79:48]=tensor_dim0, [111:80]=tensor_dim1, [127:112]=tile_dim0
    g1.w[1] = (tensor_d0 & 0xFFFFu) << 16;                       // [47:32]=atomic_barrier=0
    g1.w[2] = (tensor_d0 >> 16) | ((tensor_d1 & 0xFFFFu) << 16);
    g1.w[3] = (tensor_d1 >> 16) | (elems_per_row << 16);
    // [143:128]=tile_dim1, [159:144]=tile_dim2=0
    g1.w[4] = rows & 0xFFFFu;
    // [207:160]=tensor_dim0_stride (48b), [255:208]=tensor_dim1_stride=0
    g1.w[5] = row_stride_el;
    g1.w[6] = 0;
    g1.w[7] = 0;
    asm volatile("tensor_load_to_lds %0, %1" :: "s"(g0.v), "s"(g1.v) : "memory");
}

// ---------------------------------------------------------------- LayerNorm
__global__ __launch_bounds__(256) void ln_kernel(const float* __restrict__ x,
                                                 const float* __restrict__ g,
                                                 const float* __restrict__ b,
                                                 u16* __restrict__ out) {
    __shared__ float red[256];
    int tok = blockIdx.x, t = threadIdx.x;
    const float* xr = x + (size_t)tok * EE;
    float v0[4];
    float s = 0.f;
#pragma unroll
    for (int i = 0; i < 4; ++i) { v0[i] = xr[t + 256 * i]; s += v0[i]; }
    red[t] = s; __syncthreads();
    for (int st = 128; st > 0; st >>= 1) { if (t < st) red[t] += red[t + st]; __syncthreads(); }
    float mu = red[0] * (1.0f / EE);
    __syncthreads();
    s = 0.f;
#pragma unroll
    for (int i = 0; i < 4; ++i) { float d = v0[i] - mu; s += d * d; }
    red[t] = s; __syncthreads();
    for (int st = 128; st > 0; st >>= 1) { if (t < st) red[t] += red[t + st]; __syncthreads(); }
    float rstd = rsqrtf(red[0] * (1.0f / EE) + 1e-5f);
    u16* orow = out + (size_t)tok * EE;
#pragma unroll
    for (int i = 0; i < 4; ++i) {
        int c = t + 256 * i;
        orow[c] = f2bf((v0[i] - mu) * rstd * g[c] + b[c]);
    }
}

// ---------------------------------------------------------------- WMMA GEMM
// C[M,N] = A_bf16[M,K] @ Bw_f32[K,N] + bias (+ resid). 128x128x32 block tile,
// 256 threads = 8 waves (4 M x 2 N), each wave 32x64 = 2x4 wmma tiles.
// A tiles DMA'd by the Tensor Data Mover (wave 0), B tiles converted fp32->bf16
// and transposed to [n][k] by all lanes; double-buffered LDS, one barrier/step.
template <bool OUT_BF16, bool RESIDUAL>
__global__ __launch_bounds__(256) void gemm_kernel(const u16* __restrict__ A,
                                                   const float* __restrict__ Bw,
                                                   const float* __restrict__ bias,
                                                   const float* __restrict__ resid,
                                                   void* __restrict__ outv,
                                                   int M, int N, int K) {
    __shared__ __align__(16) u16 As[2][128 * 32];   // [m][k]
    __shared__ __align__(16) u16 Bs[2][128 * 32];   // [n][k]
    int t = threadIdx.x;
    int w = t >> 5, lane = t & 31;
    int n15 = lane & 15, half = lane >> 4;
    int wm = w & 3, wn = w >> 2;
    int mBase = blockIdx.y * 128, nBase = blockIdx.x * 128;

    v8f acc[2][4];
#pragma unroll
    for (int i = 0; i < 2; ++i)
#pragma unroll
        for (int j = 0; j < 4; ++j) acc[i][j] = vzero8();

    int bk = t >> 3, bn = (t & 7) * 16;

    // ---- prologue: stage k-block 0 into buffer 0
    if (w == 0)
        tdm_load_2d_bf16((u32)(uintptr_t)&As[0][0], A + (size_t)mBase * K,
                         32, 128, (u32)K, (u32)K, 128);
    {
        const float* src = Bw + (size_t)bk * N + nBase + bn;
#pragma unroll
        for (int i = 0; i < 16; ++i) Bs[0][(bn + i) * 32 + bk] = f2bf(src[i]);
    }
    if (w == 0) __builtin_amdgcn_s_wait_tensorcnt(0);
    __syncthreads();

    int buf = 0;
    for (int kb = 0; kb < K; kb += 32) {
        int kn = kb + 32;
        if (kn < K) {   // stage next k-block into the other buffer
            if (w == 0)
                tdm_load_2d_bf16((u32)(uintptr_t)&As[buf ^ 1][0],
                                 A + (size_t)mBase * K + kn,
                                 32, 128, (u32)K, (u32)K, 128);
            const float* src = Bw + (size_t)(kn + bk) * N + nBase + bn;
#pragma unroll
            for (int i = 0; i < 16; ++i) Bs[buf ^ 1][(bn + i) * 32 + bk] = f2bf(src[i]);
            if (kn + 32 < K)   // pull k+2 fp32 weight tile into GL2
                __builtin_prefetch(Bw + (size_t)(kn + 32 + bk) * N + nBase + bn, 0, 1);
        }

        // ---- compute on current buffer
        v16bf afr[2], bfr[4];
        afr[0] = load_a_frag(&As[buf][(wm * 32) * 32], 32);
        afr[1] = load_a_frag(&As[buf][(wm * 32 + 16) * 32], 32);
#pragma unroll
        for (int j = 0; j < 4; ++j)
            bfr[j] = load_b_frag(&Bs[buf][(wn * 64 + j * 16) * 32], 32);
#pragma unroll
        for (int i = 0; i < 2; ++i)
#pragma unroll
            for (int j = 0; j < 4; ++j) acc[i][j] = wmma_bf16(afr[i], bfr[j], acc[i][j]);

        if (w == 0) __builtin_amdgcn_s_wait_tensorcnt(0);
        __syncthreads();
        buf ^= 1;
    }

#pragma unroll
    for (int i = 0; i < 2; ++i)
#pragma unroll
        for (int j = 0; j < 4; ++j) {
            int gn = nBase + wn * 64 + j * 16 + n15;
            float bia = bias[gn];
#pragma unroll
            for (int r = 0; r < 8; ++r) {
                int gm = mBase + wm * 32 + i * 16 + r + 8 * half;
                float v = acc[i][j][r] + bia;
                if (RESIDUAL) v += resid[(size_t)gm * N + gn];
                if (OUT_BF16) ((u16*)outv)[(size_t)gm * N + gn] = f2bf(v);
                else          ((float*)outv)[(size_t)gm * N + gn] = v;
            }
        }
}

// ---------------------------------------------------------------- Attention
// Flash-attention. Block = 128 threads (4 waves), handles 64 consecutive
// queries of one (batch, head). K tile (32 keys x 64d, row stride 3072) DMA'd
// by TDM; V staged transposed so attn@V B-fragments are contiguous b128 reads.
__global__ __launch_bounds__(128) void attn_kernel(const u16* __restrict__ qkv,
                                                   u16* __restrict__ attn) {
    __shared__ __align__(16) u16 Ks[32 * 64];      // [key][d]
    __shared__ __align__(16) u16 Vt[64 * 32];      // [d][key]
    __shared__ __align__(16) u16 Ps[4 * 16 * 32];  // per-wave P tile [m][key]

    int bq = blockIdx.x;            // query chunk of 64
    int h  = blockIdx.y;
    int bz = blockIdx.z;
    int t = threadIdx.x;
    int w = t >> 5, lane = t & 31;
    int n = lane & 15, half = lane >> 4;
    size_t tokBase = (size_t)bz * TT;
    int q0 = bq * 64 + w * 16;      // this wave's first query row

    // Q A-fragments (16 x 64, split into two K=32 fragments), loaded once
    const u16* qbase = qkv + (tokBase + q0) * E3 + h * DHH;
    v16bf aq0 = load_a_frag(qbase, E3);
    v16bf aq1 = load_a_frag(qbase + 32, E3);

    float rmax[8], rsum[8];
#pragma unroll
    for (int r = 0; r < 8; ++r) { rmax[r] = -1e30f; rsum[r] = 0.0f; }
    v8f o[4];
#pragma unroll
    for (int j = 0; j < 4; ++j) o[j] = vzero8();

    int kEnd = bq * 64 + 64;        // causal bound for this block
    for (int kb = 0; kb < kEnd; kb += 32) {
        __syncthreads();
        // stage K via TDM (32 rows x 64 elems, row stride 3072 elems)
        if (w == 0)
            tdm_load_2d_bf16((u32)(uintptr_t)&Ks[0],
                             qkv + (tokBase + kb) * E3 + EE + h * DHH,
                             64, 32, E3, E3, 32);
        {   // stage V transposed (needs transpose, so cooperative VALU path)
            int key = t >> 2;                  // 0..31
            int dc  = (t & 3) * 16;            // 0,16,32,48
            const u16* vsrc = qkv + (tokBase + kb + key) * E3 + 2 * EE + h * DHH + dc;
#pragma unroll
            for (int i = 0; i < 16; ++i) Vt[(dc + i) * 32 + key] = vsrc[i];
        }
        if (w == 0) __builtin_amdgcn_s_wait_tensorcnt(0);
        __syncthreads();

        // scores S = Q @ K^T : two 16x16 key sub-tiles, K-dim 64 = 2 wmma each
        v8f s0 = vzero8(), s1 = vzero8();
        s0 = wmma_bf16(aq0, load_b_frag(&Ks[0], 64), s0);
        s0 = wmma_bf16(aq1, load_b_frag(&Ks[32], 64), s0);
        s1 = wmma_bf16(aq0, load_b_frag(&Ks[16 * 64], 64), s1);
        s1 = wmma_bf16(aq1, load_b_frag(&Ks[16 * 64 + 32], 64), s1);

        // online softmax (fp32); row m = r + 8*half, key col = kb + n (+16)
        u16* P = &Ps[w * 512];
        float al[8];
#pragma unroll
        for (int r = 0; r < 8; ++r) {
            int m = r + 8 * half;
            int q = q0 + m;
            float v0 = (kb + n > q)      ? -1e30f : s0[r] * 0.125f;  // DH^-0.5
            float v1 = (kb + 16 + n > q) ? -1e30f : s1[r] * 0.125f;
            float mx = red_max16(fmaxf(v0, v1));      // ds_swizzle butterflies
            float nm = fmaxf(rmax[r], mx);
            float a  = __expf(rmax[r] - nm);
            float p0 = __expf(v0 - nm), p1 = __expf(v1 - nm);
            float ls = red_sum16(p0 + p1);
            rsum[r] = rsum[r] * a + ls;
            rmax[r] = nm;
            al[r]   = a;
            P[m * 32 + n]      = f2bf(p0);
            P[m * 32 + 16 + n] = f2bf(p1);
        }
        __syncthreads();   // P C->A relayout fence (uniform across block)

        v16bf aP = load_a_frag(P, 32);
#pragma unroll
        for (int j = 0; j < 4; ++j) {
#pragma unroll
            for (int r = 0; r < 8; ++r) o[j][r] *= al[r];   // rescale, then add
            o[j] = wmma_bf16(aP, load_b_frag(&Vt[j * 16 * 32], 32), o[j]);
        }
    }

    // epilogue: O /= rowsum, write bf16 [tok][h*64 + d]
#pragma unroll
    for (int j = 0; j < 4; ++j) {
        int d = h * DHH + j * 16 + n;
#pragma unroll
        for (int r = 0; r < 8; ++r) {
            int m = r + 8 * half;
            size_t tok = tokBase + q0 + m;
            attn[tok * EE + d] = f2bf(o[j][r] / rsum[r]);
        }
    }
}

// ---------------------------------------------------------------- Argmax
__global__ __launch_bounds__(256) void argmax_kernel(const float* __restrict__ logits,
                                                     int* __restrict__ idx) {
    __shared__ float sv[256];
    __shared__ int   si[256];
    int tok = blockIdx.x, t = threadIdx.x;
    const float* lr = logits + (size_t)tok * NEE;
    float bv = -3.4e38f; int bi = 0;
#pragma unroll
    for (int i = 0; i < 8; ++i) {
        int c = t + 256 * i;
        float v = lr[c];
        if (v > bv) { bv = v; bi = c; }
    }
    sv[t] = bv; si[t] = bi; __syncthreads();
    for (int st = 128; st > 0; st >>= 1) {
        if (t < st) {
            if (sv[t + st] > sv[t] || (sv[t + st] == sv[t] && si[t + st] < si[t])) {
                sv[t] = sv[t + st]; si[t] = si[t + st];
            }
        }
        __syncthreads();
    }
    if (t == 0) idx[tok] = si[0];
}

// ---------------------------------------------------------------- MoE (top-1)
__global__ __launch_bounds__(256) void moe_kernel(const u16* __restrict__ h2,
                                                  const int* __restrict__ idx,
                                                  const float* __restrict__ w1,
                                                  const float* __restrict__ b1,
                                                  const float* __restrict__ w2,
                                                  const float* __restrict__ b2,
                                                  const float* __restrict__ x1,
                                                  float* __restrict__ out) {
    __shared__ float red[256 * HIDD];
    __shared__ float hm[HIDD];
    int tok = blockIdx.x, t = threadIdx.x;
    int e = idx[tok];
    const float* W1 = w1 + (size_t)e * EE * HIDD;

    float acc[HIDD];
#pragma unroll
    for (int hh = 0; hh < HIDD; ++hh) acc[hh] = 0.0f;
#pragma unroll
    for (int i = 0; i < 4; ++i) {
        int k = t + 256 * i;
        float xv = bf2f(h2[(size_t)tok * EE + k]);
        const float* r = W1 + (size_t)k * HIDD;
#pragma unroll
        for (int hh = 0; hh < HIDD; ++hh) acc[hh] += xv * r[hh];
    }
#pragma unroll
    for (int hh = 0; hh < HIDD; ++hh) red[t * HIDD + hh] = acc[hh];
    __syncthreads();
    for (int st = 128; st > 0; st >>= 1) {
        if (t < st)
#pragma unroll
            for (int hh = 0; hh < HIDD; ++hh) red[t * HIDD + hh] += red[(t + st) * HIDD + hh];
        __syncthreads();
    }
    if (t < HIDD) {
        float v = red[t] + b1[(size_t)e * HIDD + t];
        hm[t] = 0.5f * v * (1.0f + erff(v * 0.70710678118654752f));   // exact gelu
    }
    __syncthreads();

    const float* W2 = w2 + (size_t)e * HIDD * EE;
#pragma unroll
    for (int i = 0; i < 4; ++i) {
        int eo = t + 256 * i;
        float y = b2[(size_t)e * EE + eo];
#pragma unroll
        for (int hh = 0; hh < HIDD; ++hh) y += hm[hh] * W2[hh * EE + eo];
        out[(size_t)tok * EE + eo] = x1[(size_t)tok * EE + eo] + y;
    }
}

// ---------------------------------------------------------------- launch
extern "C" void kernel_launch(void* const* d_in, const int* in_sizes, int n_in,
                              void* d_out, int out_size, void* d_ws, size_t ws_size,
                              hipStream_t stream) {
    (void)in_sizes; (void)n_in; (void)out_size; (void)ws_size;
    const float* x      = (const float*)d_in[0];
    const float* ln1_g  = (const float*)d_in[1];
    const float* ln1_b  = (const float*)d_in[2];
    const float* ln2_g  = (const float*)d_in[3];
    const float* ln2_b  = (const float*)d_in[4];
    const float* qkv_w  = (const float*)d_in[5];
    const float* qkv_b  = (const float*)d_in[6];
    const float* proj_w = (const float*)d_in[7];
    const float* proj_b = (const float*)d_in[8];
    const float* gate_w = (const float*)d_in[9];
    const float* gate_b = (const float*)d_in[10];
    const float* w1     = (const float*)d_in[11];
    const float* b1     = (const float*)d_in[12];
    const float* w2     = (const float*)d_in[13];
    const float* b2     = (const float*)d_in[14];
    float* out = (float*)d_out;

    char* ws = (char*)d_ws;
    size_t off = 0;
    u16*   h1     = (u16*)(ws + off);   off += (size_t)MTOK * EE * 2;     // 16 MB
    u16*   qkvb   = (u16*)(ws + off);   off += (size_t)MTOK * E3 * 2;     // 48 MB
    u16*   attnb  = (u16*)(ws + off);   off += (size_t)MTOK * EE * 2;     // 16 MB
    float* x1     = (float*)(ws + off); off += (size_t)MTOK * EE * 4;     // 32 MB
    u16*   h2     = (u16*)(ws + off);   off += (size_t)MTOK * EE * 2;     // 16 MB
    float* logits = (float*)(ws + off); off += (size_t)MTOK * NEE * 4;    // 64 MB
    int*   eidx   = (int*)(ws + off);   off += (size_t)MTOK * 4;

    // 1. h1 = ln1(x)  (bf16)
    ln_kernel<<<MTOK, 256, 0, stream>>>(x, ln1_g, ln1_b, h1);
    // 2. qkv = h1 @ qkv_w + qkv_b  (bf16 out)
    gemm_kernel<true, false><<<dim3(E3 / 128, MTOK / 128), 256, 0, stream>>>(
        h1, qkv_w, qkv_b, nullptr, qkvb, MTOK, E3, EE);
    // 3. flash attention
    attn_kernel<<<dim3(TT / 64, NHH, BB), 128, 0, stream>>>(qkvb, attnb);
    // 4. x1 = x + attn @ proj_w + proj_b  (fp32)
    gemm_kernel<false, true><<<dim3(EE / 128, MTOK / 128), 256, 0, stream>>>(
        attnb, proj_w, proj_b, x, x1, MTOK, EE, EE);
    // 5. h2 = ln2(x1)  (bf16)
    ln_kernel<<<MTOK, 256, 0, stream>>>(x1, ln2_g, ln2_b, h2);
    // 6. logits = h2 @ gate_w + gate_b  (fp32)
    gemm_kernel<false, false><<<dim3(NEE / 128, MTOK / 128), 256, 0, stream>>>(
        h2, gate_w, gate_b, nullptr, logits, MTOK, NEE, EE);
    // 7. expert index
    argmax_kernel<<<MTOK, 256, 0, stream>>>(logits, eidx);
    // 8. out = x1 + moe(h2)
    moe_kernel<<<MTOK, 256, 0, stream>>>(h2, eidx, w1, b1, w2, b2, x1, out);
}
